// DebiasIntraDist_84052509982804
// MI455X (gfx1250) — compile-verified, matrix operand order checked by reading, not run on hardware
//
#include <hip/hip_runtime.h>
#include <hip/hip_bf16.h>

#define NUM_LABELS 512
#define NUM_DEMOG  4
#define G          (NUM_DEMOG * NUM_LABELS)   // 2048
#define D          512

typedef __attribute__((ext_vector_type(16))) _Float16 v16h;
typedef __attribute__((ext_vector_type(8)))  float    v8f;

// ---------------------------------------------------------------------------
// Kernel 0: zero the workspace accumulators.
// ---------------------------------------------------------------------------
__global__ void k_zero(float* __restrict__ p, int n) {
    int i = blockIdx.x * blockDim.x + threadIdx.x;
    if (i < n) p[i] = 0.0f;
}

// ---------------------------------------------------------------------------
// Kernel 1: streaming pass over feats.  One wave32 per row-iteration.
//   - 128-bit coalesced loads (512 B per wave step)
//   - global_atomic_add_f32 into L2-resident sums[G][D]
//   - wave-reduced sum-of-squares, single atomic per row
// ---------------------------------------------------------------------------
__global__ void k_accumulate(const float* __restrict__ feats,
                             const int*  __restrict__ labels,
                             const int*  __restrict__ demog,
                             float* __restrict__ sums,
                             float* __restrict__ ssq,
                             float* __restrict__ cntf,
                             int n) {
    const int lane   = threadIdx.x & 31;
    const int wave   = (blockIdx.x * blockDim.x + threadIdx.x) >> 5;
    const int nwaves = (gridDim.x * blockDim.x) >> 5;

    for (int r = wave; r < n; r += nwaves) {
        const int seg = demog[r] * NUM_LABELS + labels[r];
        const float* __restrict__ rowp = feats + (size_t)r * D;
        float* __restrict__ srow       = sums  + (size_t)seg * D;

        // Prefetch the next row this wave will process (global_prefetch_b8).
        if (r + nwaves < n)
            __builtin_prefetch(feats + (size_t)(r + nwaves) * D + lane * 16, 0, 1);

        float part = 0.0f;
#pragma unroll
        for (int i = 0; i < 4; ++i) {
            const int idx = i * 128 + lane * 4;
            const float4 v = *(const float4*)(rowp + idx);
            part += v.x * v.x + v.y * v.y + v.z * v.z + v.w * v.w;
            unsafeAtomicAdd(&srow[idx + 0], v.x);
            unsafeAtomicAdd(&srow[idx + 1], v.y);
            unsafeAtomicAdd(&srow[idx + 2], v.z);
            unsafeAtomicAdd(&srow[idx + 3], v.w);
        }
        // wave32 butterfly reduction of the squared-norm partial
#pragma unroll
        for (int m = 16; m >= 1; m >>= 1)
            part += __shfl_xor(part, m, 32);
        if (lane == 0) {
            unsafeAtomicAdd(&ssq[seg], part);
            unsafeAtomicAdd(&cntf[seg], 1.0f);
        }
    }
}

// ---------------------------------------------------------------------------
// Kernel 2: ||sums_g||^2 via WMMA.  One wave per 16-group tile; accumulate
// the 16x16 gram D = S * S^T over 16 K-chunks of 32 (K total = 512) with
// v_wmma_f32_16x16x32_f16, then read off the diagonal.
// Feeding identical registers as A and B makes the diagonal exact w.r.t.
// the per-lane element pairing regardless of intra-chunk K ordering.
// ---------------------------------------------------------------------------
__global__ void k_groupnorm(const float* __restrict__ sums,
                            float* __restrict__ norm2) {
    const int tile = blockIdx.x;          // G/16 = 128 tiles
    const int lane = threadIdx.x;         // 32 threads = 1 wave
    const int m    = lane & 15;           // group row within tile
    // lanes 0-15 take halves [0,16) of each 32-wide K chunk, lanes 16-31 take [16,32)
    const float* __restrict__ row =
        sums + (size_t)(tile * 16 + m) * D + ((lane >> 4) << 4);

    v8f c = {};
    for (int t = 0; t < 16; ++t) {
        const float* __restrict__ p = row + t * 32;
        v16h a;
#pragma unroll
        for (int j = 0; j < 16; ++j) a[j] = (_Float16)p[j];
        // (neg_a, A, neg_b, B, c_mod, C, reuse_a, reuse_b)
        c = __builtin_amdgcn_wmma_f32_16x16x32_f16(
                false, a, false, a, (short)0, c, false, false);
    }

    // Diagonal (g,g): m<8 -> lane m, elem m ; m>=8 -> lane m+16, elem m-8.
    float val = 0.0f;
    int   mm  = -1;
    if (lane < 8)        mm = lane;
    else if (lane >= 24) mm = lane - 16;
#pragma unroll
    for (int j = 0; j < 8; ++j) {
        const bool mine = (lane < 8 && j == lane) || (lane >= 24 && j == lane - 24);
        val = mine ? c[j] : val;
    }
    if (mm >= 0) norm2[tile * 16 + mm] = val;
}

// ---------------------------------------------------------------------------
// Kernel 3: finalize.  gm_g = (ssq_g - ||sums_g||^2 / cnt) / cnt for present
// groups; per-demog means via LDS float atomics (ds_add_f32); scalar loss.
// ---------------------------------------------------------------------------
__global__ void k_finalize(const float* __restrict__ ssq,
                           const float* __restrict__ cntf,
                           const float* __restrict__ norm2,
                           float* __restrict__ out) {
    __shared__ float sgm[NUM_DEMOG];
    __shared__ float spres[NUM_DEMOG];
    const int tid = threadIdx.x;
    if (tid < NUM_DEMOG) { sgm[tid] = 0.0f; spres[tid] = 0.0f; }
    __syncthreads();

    for (int g = tid; g < G; g += blockDim.x) {
        const float cn = cntf[g];
        if (cn > 0.0f) {
            const float gm = (ssq[g] - norm2[g] / cn) / cn;
            atomicAdd(&sgm[g >> 9], gm);      // g / NUM_LABELS
            atomicAdd(&spres[g >> 9], 1.0f);
        }
    }
    __syncthreads();

    if (tid == 0) {
        float intra[NUM_DEMOG];
        float mu = 0.0f;
        for (int d = 0; d < NUM_DEMOG; ++d) {
            intra[d] = sgm[d] / fmaxf(spres[d], 1.0f);
            mu += intra[d];
        }
        mu *= (1.0f / NUM_DEMOG);
        float loss = 0.0f;
        for (int d = 0; d < NUM_DEMOG; ++d) loss += fabsf(intra[d] - mu);
        out[0] = loss * (1.0f / NUM_DEMOG);
    }
}

// ---------------------------------------------------------------------------
extern "C" void kernel_launch(void* const* d_in, const int* in_sizes, int n_in,
                              void* d_out, int out_size, void* d_ws, size_t ws_size,
                              hipStream_t stream) {
    const float* feats  = (const float*)d_in[0];
    const int*   labels = (const int*)d_in[1];
    const int*   demog  = (const int*)d_in[2];
    float*       out    = (float*)d_out;

    float* ws    = (float*)d_ws;
    float* sums  = ws;                       // G*D floats (4 MB)
    float* ssq   = sums + (size_t)G * D;     // G floats
    float* cntf  = ssq + G;                  // G floats
    float* norm2 = cntf + G;                 // G floats

    const int n = in_sizes[1];               // N rows

    const int ztotal = G * D + 3 * G;
    k_zero<<<(ztotal + 255) / 256, 256, 0, stream>>>(ws, ztotal);

    k_accumulate<<<1024, 256, 0, stream>>>(feats, labels, demog,
                                           sums, ssq, cntf, n);

    k_groupnorm<<<G / 16, 32, 0, stream>>>(sums, norm2);

    k_finalize<<<1, 256, 0, stream>>>(ssq, cntf, norm2, out);
}
